// LSTMPolicy_88940182765653
// MI455X (gfx1250) — matrix operand order repaired and checked by reading, not compiled, not run on hardware
//
#include <hip/hip_runtime.h>
#include <hip/hip_bf16.h>

typedef __attribute__((ext_vector_type(16))) _Float16 v16h;
typedef __attribute__((ext_vector_type(8)))  _Float16 v8h;
typedef __attribute__((ext_vector_type(8)))  float    v8f;

#define HIDDEN 128
#define INSZ   64
#define NACT   16
#define BATCHN 65536

// Packed weight sizes, in halves. 3 live gates (i,g,o) x 8 n-chunks = 24 tiles.
// Each tile-kchunk: 32 lanes x 16 halves = 512 halves.
#define W0_HALVES (24 * 2 * 512)   // layer0: K=64  -> 2 kchunks
#define W1_HALVES (24 * 4 * 512)   // layer1: K=128 -> 4 kchunks
#define WP_HALVES (1 * 4 * 512)    // policy head: 1 tile, 4 kchunks
#define WV_HALVES (1 * 4 * 512)    // value head (padded to 16 cols)

// A-matrix 16x32 f16 K mapping (ISA 7.12.2): lanes g=lane/16,
//   j in [0,8): k = j + 8g ; j in [8,16): k = 16 + (j-8) + 8g
__device__ __forceinline__ int kmapA(int j, int lane, int kc) {
  return kc * 32 + ((j < 8) ? j : j + 8) + ((lane >> 4) << 3);
}
// B-matrix 32x16 f16 K mapping, same as A: g=lane/16,
//   j in [0,8): k = j + 8g ; j in [8,16): k = 16 + (j-8) + 8g
__device__ __forceinline__ int kmapB(int j, int lane, int kc) {
  return kc * 32 + ((j < 8) ? j : j + 8) + ((lane >> 4) << 3);
}

__device__ __forceinline__ float sigmoid_fast(float x) {
  return 1.0f / (1.0f + __expf(-x));
}
__device__ __forceinline__ float tanh_fast(float x) {
  float ax = fabsf(x);
  float t  = __expf(-2.0f * ax);
  float r  = (1.0f - t) / (1.0f + t);
  return copysignf(r, x);
}

// ---------------------------------------------------------------------------
// Prepack: f32 weights -> f16 WMMA B-fragments in workspace; fuse bih+bhh.
// Gate row offsets (i,f,g,o): we keep only i(0), g(2H), o(3H); f is dead (c=0).
// ---------------------------------------------------------------------------
__global__ void lstm_prepack(const float* __restrict__ Wih0,
                             const float* __restrict__ bih0, const float* __restrict__ bhh0,
                             const float* __restrict__ Wih1,
                             const float* __restrict__ bih1, const float* __restrict__ bhh1,
                             const float* __restrict__ Wp,  const float* __restrict__ bp,
                             const float* __restrict__ Wv,  const float* __restrict__ bv,
                             _Float16* __restrict__ pW0, _Float16* __restrict__ pW1,
                             _Float16* __restrict__ pWp, _Float16* __restrict__ pWv,
                             float* __restrict__ b0, float* __restrict__ b1,
                             float* __restrict__ bpo, float* __restrict__ bvo) {
  int e = blockIdx.x * blockDim.x + threadIdx.x;
  const int goff[3] = {0, 2 * HIDDEN, 3 * HIDDEN};

  if (e < W0_HALVES) {
    int j = e & 15, lane = (e >> 4) & 31, kc = (e >> 9) & 1, t = e >> 10;
    int grp = t >> 3, nc = t & 7;
    int n = goff[grp] + nc * 16 + (lane & 15);
    int k = kmapB(j, lane, kc);
    pW0[e] = (_Float16)Wih0[n * INSZ + k];
    return;
  }
  e -= W0_HALVES;
  if (e < W1_HALVES) {
    int j = e & 15, lane = (e >> 4) & 31, kc = (e >> 9) & 3, t = e >> 11;
    int grp = t >> 3, nc = t & 7;
    int n = goff[grp] + nc * 16 + (lane & 15);
    int k = kmapB(j, lane, kc);
    pW1[e] = (_Float16)Wih1[n * HIDDEN + k];
    return;
  }
  e -= W1_HALVES;
  if (e < WP_HALVES) {
    int j = e & 15, lane = (e >> 4) & 31, kc = (e >> 9) & 3;
    int n = lane & 15;
    int k = kmapB(j, lane, kc);
    pWp[e] = (_Float16)Wp[n * HIDDEN + k];
    return;
  }
  e -= WP_HALVES;
  if (e < WV_HALVES) {
    int j = e & 15, lane = (e >> 4) & 31, kc = (e >> 9) & 3;
    int n = lane & 15;
    int k = kmapB(j, lane, kc);
    pWv[e] = (n == 0) ? (_Float16)Wv[k] : (_Float16)0.0f;
    return;
  }
  e -= WV_HALVES;
  if (e < 3 * HIDDEN) {
    int grp = e >> 7, h = e & 127;
    int n = goff[grp] + h;
    b0[e] = bih0[n] + bhh0[n];
    return;
  }
  e -= 3 * HIDDEN;
  if (e < 3 * HIDDEN) {
    int grp = e >> 7, h = e & 127;
    int n = goff[grp] + h;
    b1[e] = bih1[n] + bhh1[n];
    return;
  }
  e -= 3 * HIDDEN;
  if (e < 16) { bpo[e] = bp[e]; return; }
  e -= 16;
  if (e < 16) { bvo[e] = (e == 0) ? bv[0] : 0.0f; return; }
}

// ---------------------------------------------------------------------------
// Fused 2-layer LSTM step + heads. 256 threads = 8 waves; wave = 16 batch rows.
// ---------------------------------------------------------------------------
__global__ __launch_bounds__(256, 1)
void lstm_fused(const float* __restrict__ x,
                const _Float16* __restrict__ pW0, const _Float16* __restrict__ pW1,
                const _Float16* __restrict__ pWp, const _Float16* __restrict__ pWv,
                const float* __restrict__ b0, const float* __restrict__ b1,
                const float* __restrict__ bpo, const float* __restrict__ bvo,
                float* __restrict__ out) {
  __shared__ _Float16 ldsH0[8][16][HIDDEN];  // per-wave h0 staging (D->A relayout)
  __shared__ _Float16 ldsH1[8][16][HIDDEN];  // per-wave h1 staging

  const int lane = threadIdx.x & 31;
  const int wave = threadIdx.x >> 5;
  const int m    = lane & 15;            // A-row / C-column owned by this lane
  const int g8   = (lane >> 4) << 3;     // A-fragment K group offset
  const int rowBase = (blockIdx.x * 8 + wave) * 16;
  const int row  = rowBase + m;

  // ---- A0 fragments from x[row], K=64 -> 2 kchunks of 32 (f32->f16 convert)
  v16h a0[2];
#pragma unroll
  for (int kc = 0; kc < 2; ++kc) {
    const float* xr = x + (size_t)row * INSZ + kc * 32 + g8;
    const float4 s0 = *(const float4*)(xr + 0);
    const float4 s1 = *(const float4*)(xr + 4);
    const float4 s2 = *(const float4*)(xr + 16);
    const float4 s3 = *(const float4*)(xr + 20);
    v16h a;
    a[0] = (_Float16)s0.x;  a[1] = (_Float16)s0.y;  a[2] = (_Float16)s0.z;  a[3] = (_Float16)s0.w;
    a[4] = (_Float16)s1.x;  a[5] = (_Float16)s1.y;  a[6] = (_Float16)s1.z;  a[7] = (_Float16)s1.w;
    a[8] = (_Float16)s2.x;  a[9] = (_Float16)s2.y;  a[10] = (_Float16)s2.z; a[11] = (_Float16)s2.w;
    a[12] = (_Float16)s3.x; a[13] = (_Float16)s3.y; a[14] = (_Float16)s3.z; a[15] = (_Float16)s3.w;
    a0[kc] = a;
  }

  // ---- Layer 0: per 16-wide hidden chunk, gates i/g/o; f-gate dead (c_prev=0)
  for (int nc = 0; nc < 8; ++nc) {
    v8f acc[3];
#pragma unroll
    for (int g = 0; g < 3; ++g) {
      const float bias = b0[g * HIDDEN + nc * 16 + m];
      v8f c;
#pragma unroll
      for (int r = 0; r < 8; ++r) c[r] = bias;
#pragma unroll
      for (int kc = 0; kc < 2; ++kc) {
        const v16h b = *(const v16h*)(pW0 + (size_t)(((g * 8 + nc) * 2 + kc) * 32 + lane) * 16);
        c = __builtin_amdgcn_wmma_f32_16x16x32_f16(false, a0[kc], false, b,
                                                   (short)0, c, false, false);
      }
      acc[g] = c;
    }
#pragma unroll
    for (int r = 0; r < 8; ++r) {
      const float ig = sigmoid_fast(acc[0][r]);
      const float gg = tanh_fast(acc[1][r]);
      const float og = sigmoid_fast(acc[2][r]);
      const float h  = og * tanh_fast(ig * gg);
      const int   M  = r + ((lane >> 4) << 3);   // D layout: row
      ldsH0[wave][M][nc * 16 + m] = (_Float16)h; // column = lane%16
    }
  }

  // ---- A1 fragments for layer 1 from LDS (K=128 -> 4 kchunks). Per-wave
  // private region; DS ops are in-order within a wave, no barrier needed.
  v16h a1[4];
#pragma unroll
  for (int kc = 0; kc < 4; ++kc) {
    const _Float16* hp = &ldsH0[wave][m][kc * 32 + g8];
    const v8h lo = *(const v8h*)(hp);
    const v8h hi = *(const v8h*)(hp + 16);
    v16h a;
#pragma unroll
    for (int j = 0; j < 8; ++j) { a[j] = lo[j]; a[j + 8] = hi[j]; }
    a1[kc] = a;
  }

  // ---- Layer 1
  for (int nc = 0; nc < 8; ++nc) {
    v8f acc[3];
#pragma unroll
    for (int g = 0; g < 3; ++g) {
      const float bias = b1[g * HIDDEN + nc * 16 + m];
      v8f c;
#pragma unroll
      for (int r = 0; r < 8; ++r) c[r] = bias;
#pragma unroll
      for (int kc = 0; kc < 4; ++kc) {
        const v16h b = *(const v16h*)(pW1 + (size_t)(((g * 8 + nc) * 4 + kc) * 32 + lane) * 16);
        c = __builtin_amdgcn_wmma_f32_16x16x32_f16(false, a1[kc], false, b,
                                                   (short)0, c, false, false);
      }
      acc[g] = c;
    }
#pragma unroll
    for (int r = 0; r < 8; ++r) {
      const float ig = sigmoid_fast(acc[0][r]);
      const float gg = tanh_fast(acc[1][r]);
      const float og = sigmoid_fast(acc[2][r]);
      const float h  = og * tanh_fast(ig * gg);
      const int   M  = r + ((lane >> 4) << 3);
      ldsH1[wave][M][nc * 16 + m] = (_Float16)h;
    }
  }

  // ---- h1 A-fragments
  v16h ah[4];
#pragma unroll
  for (int kc = 0; kc < 4; ++kc) {
    const _Float16* hp = &ldsH1[wave][m][kc * 32 + g8];
    const v8h lo = *(const v8h*)(hp);
    const v8h hi = *(const v8h*)(hp + 16);
    v16h a;
#pragma unroll
    for (int j = 0; j < 8; ++j) { a[j] = lo[j]; a[j + 8] = hi[j]; }
    ah[kc] = a;
  }

  // ---- Heads: policy [16x16], value [16x1 padded to 16]
  v8f accP, accV;
  {
    const float biasP = bpo[m];
    const float biasV = bvo[m];
#pragma unroll
    for (int r = 0; r < 8; ++r) { accP[r] = biasP; accV[r] = biasV; }
#pragma unroll
    for (int kc = 0; kc < 4; ++kc) {
      const v16h bP = *(const v16h*)(pWp + (size_t)(kc * 32 + lane) * 16);
      accP = __builtin_amdgcn_wmma_f32_16x16x32_f16(false, ah[kc], false, bP,
                                                    (short)0, accP, false, false);
      const v16h bV = *(const v16h*)(pWv + (size_t)(kc * 32 + lane) * 16);
      accV = __builtin_amdgcn_wmma_f32_16x16x32_f16(false, ah[kc], false, bV,
                                                    (short)0, accV, false, false);
    }
  }

  // ---- Store: policy [B,16] row-major, then value [B] appended.
#pragma unroll
  for (int r = 0; r < 8; ++r) {
    const int M = r + ((lane >> 4) << 3);
    out[(size_t)(rowBase + M) * NACT + m] = accP[r];
  }
  if (m == 0) {  // lanes 0 (rows 0-7) and 16 (rows 8-15) hold value column 0
#pragma unroll
    for (int r = 0; r < 8; ++r) {
      const int M = r + ((lane >> 4) << 3);
      out[(size_t)BATCHN * NACT + rowBase + M] = accV[r];
    }
  }
}

extern "C" void kernel_launch(void* const* d_in, const int* in_sizes, int n_in,
                              void* d_out, int out_size, void* d_ws, size_t ws_size,
                              hipStream_t stream) {
  const float* x    = (const float*)d_in[0];
  const float* Wih0 = (const float*)d_in[1];
  // d_in[2] = Whh0 : dead (h_prev = 0)
  const float* bih0 = (const float*)d_in[3];
  const float* bhh0 = (const float*)d_in[4];
  const float* Wih1 = (const float*)d_in[5];
  // d_in[6] = Whh1 : dead
  const float* bih1 = (const float*)d_in[7];
  const float* bhh1 = (const float*)d_in[8];
  const float* Wp   = (const float*)d_in[9];
  const float* bp   = (const float*)d_in[10];
  const float* Wv   = (const float*)d_in[11];
  const float* bv   = (const float*)d_in[12];

  char* ws = (char*)d_ws;
  _Float16* pW0 = (_Float16*)ws; ws += (size_t)W0_HALVES * 2;
  _Float16* pW1 = (_Float16*)ws; ws += (size_t)W1_HALVES * 2;
  _Float16* pWp = (_Float16*)ws; ws += (size_t)WP_HALVES * 2;
  _Float16* pWv = (_Float16*)ws; ws += (size_t)WV_HALVES * 2;
  float* b0  = (float*)ws; ws += 3 * HIDDEN * 4;
  float* b1  = (float*)ws; ws += 3 * HIDDEN * 4;
  float* bpo = (float*)ws; ws += 16 * 4;
  float* bvo = (float*)ws; ws += 16 * 4;

  const int totalPre = W0_HALVES + W1_HALVES + WP_HALVES + WV_HALVES +
                       3 * HIDDEN + 3 * HIDDEN + 16 + 16;
  lstm_prepack<<<(totalPre + 255) / 256, 256, 0, stream>>>(
      Wih0, bih0, bhh0, Wih1, bih1, bhh1, Wp, bp, Wv, bv,
      pW0, pW1, pWp, pWv, b0, b1, bpo, bvo);

  lstm_fused<<<BATCHN / 128, 256, 0, stream>>>(
      x, pW0, pW1, pWp, pWv, b0, b1, bpo, bvo, (float*)d_out);
}
